// GAT_IoT_Classifier_77678778515804
// MI455X (gfx1250) — matrix operand order repaired and checked by reading, not compiled, not run on hardware
//
#include <hip/hip_runtime.h>
#include <math.h>

// GAT 2-layer + MLP head for MI455X (gfx1250, wave32, WMMA).
// GEMMs use v_wmma_f32_16x16x32_f16 (f16 in, f32 accumulate).
// Edge softmax/aggregation via atomics (dominant cost is HBM traffic, not FLOPs).

#define HEADS 4
#define OUTC  64
#define HC    256   // HEADS*OUTC

typedef __attribute__((ext_vector_type(16))) _Float16 v16h;
typedef __attribute__((ext_vector_type(8)))  float    v8f;

union F16Frag { uint4 q[2]; v16h v; };

// ---- monotonic uint key for float atomicMax (handles negatives) ----
__device__ __forceinline__ unsigned fkey(float f) {
  unsigned u = __float_as_uint(f);
  return (u & 0x80000000u) ? ~u : (u | 0x80000000u);
}
__device__ __forceinline__ float fdecode(unsigned k) {
  return __uint_as_float((k & 0x80000000u) ? (k ^ 0x80000000u) : ~k);
}
__device__ __forceinline__ unsigned pack2h(float a, float b) {
  union { _Float16 h[2]; unsigned u; } c;
  c.h[0] = (_Float16)a; c.h[1] = (_Float16)b;
  return c.u;
}
__device__ __forceinline__ float lrelu02(float x) { return x > 0.f ? x : 0.2f * x; }

// ---------------- elementwise f32 -> f16 ----------------
__global__ void f32_to_f16_kernel(const float* __restrict__ in, _Float16* __restrict__ out, int n) {
  int i = blockIdx.x * blockDim.x + threadIdx.x;
  if (i < n) out[i] = (_Float16)in[i];
}

// ---- pack weights W[K x 256] (f32, row-major) into WMMA B-fragment order ----
// Tile (kt, nt): 256 dwords; lane holds 8 contiguous dwords.
// dword i of lane l = half2( B[kt*32 + 16*(l>>4) + 2i][nt*16 + (l&15)],
//                            B[kt*32 + 16*(l>>4) + 2i + 1][same col] )
__global__ void pack_b_kernel(const float* __restrict__ W, unsigned* __restrict__ Bp, int K) {
  int d = blockIdx.x * blockDim.x + threadIdx.x;
  int total = (K / 32) * 16 * 256;
  if (d >= total) return;
  int tile   = d >> 8;
  int within = d & 255;
  int lane   = within >> 3;
  int i      = within & 7;
  int kt = tile >> 4;
  int nt = tile & 15;
  int k = kt * 32 + 16 * (lane >> 4) + 2 * i;
  int n = nt * 16 + (lane & 15);
  Bp[d] = pack2h(W[(size_t)k * HC + n], W[(size_t)(k + 1) * HC + n]);
}

// ---------------- WMMA GEMM: C[N x 256] = A[N x K](f16) @ Bpack ----------------
// One wave per 16x16 output tile. N must be a multiple of 16 (N=50000 ok).
__global__ __launch_bounds__(256) void gemm_wmma_kernel(const _Float16* __restrict__ A,
                                                        const unsigned* __restrict__ Bp,
                                                        float* __restrict__ C, int N, int K) {
  int wave = (blockIdx.x * blockDim.x + threadIdx.x) >> 5;
  int lane = threadIdx.x & 31;
  int nTiles = (N >> 4) * 16;
  if (wave >= nTiles) return;
  int tm = wave >> 4;
  int tn = wave & 15;
  int grp = lane >> 4;
  int r15 = lane & 15;
  const _Float16* Arow = A + (size_t)(tm * 16 + r15) * K;
  v8f acc = {};
  int KT = K >> 5;
  for (int kt = 0; kt < KT; kt++) {
    F16Frag a, b;
    // A 16x32 f16 fragment: elems 0-15 -> K = 16*grp..+15
    a.q[0] = *(const uint4*)(Arow + kt * 32 + 16 * grp);
    a.q[1] = *(const uint4*)(Arow + kt * 32 + 16 * grp + 8);
    const uint4* bp = (const uint4*)(Bp + ((size_t)(kt * 16 + tn) * 256 + lane * 8));
    b.q[0] = bp[0];
    b.q[1] = bp[1];
    acc = __builtin_amdgcn_wmma_f32_16x16x32_f16(false, a.v, false, b.v, (short)0, acc, false, false);
  }
  // C/D layout: VGPR r -> row = tm*16 + r + 8*grp, col = tn*16 + (lane&15)
  float* Crow = C + (size_t)(tm * 16 + 8 * grp) * HC + tn * 16 + r15;
#pragma unroll
  for (int r = 0; r < 8; r++) Crow[(size_t)r * HC] = acc[r];
}

// ------------- per-(node,head) attention scores: a_s, a_d [N x 4] -------------
// grid = N blocks, 128 threads (4 waves; wave w = head w)
__global__ __launch_bounds__(128) void att_score_kernel(const float* __restrict__ h,
                                                        const float* __restrict__ as_w,
                                                        const float* __restrict__ ad_w,
                                                        float* __restrict__ as_o,
                                                        float* __restrict__ ad_o) {
  int n = blockIdx.x;
  int w = threadIdx.x >> 5;
  int l = threadIdx.x & 31;
  const float* row = h + (size_t)n * HC + w * OUTC;
  float ss = row[l] * as_w[w * OUTC + l] + row[l + 32] * as_w[w * OUTC + l + 32];
  float sd = row[l] * ad_w[w * OUTC + l] + row[l + 32] * ad_w[w * OUTC + l + 32];
#pragma unroll
  for (int off = 16; off; off >>= 1) {
    ss += __shfl_down(ss, off, 32);
    sd += __shfl_down(sd, off, 32);
  }
  if (l == 0) { as_o[n * 4 + w] = ss; ad_o[n * 4 + w] = sd; }
}

__global__ void init_mkey_kernel(unsigned* __restrict__ mkey, int n) {
  int i = blockIdx.x * blockDim.x + threadIdx.x;
  if (i < n) mkey[i] = 0x007FFFFFu;  // fkey(-inf)
}

// edge pass 1: segment max of leaky_relu(a_s[src]+a_d[dst]) over dst
__global__ void edge_max_kernel(const int* __restrict__ ei, int E, int N,
                                const float* __restrict__ as_, const float* __restrict__ ad_,
                                unsigned* __restrict__ mkey) {
  int e = blockIdx.x * blockDim.x + threadIdx.x;
  if (e >= E + N) return;
  int s = (e < E) ? ei[e] : (e - E);
  int d = (e < E) ? ei[E + e] : (e - E);
#pragma unroll
  for (int hh = 0; hh < 4; hh++) {
    float lg = lrelu02(as_[s * 4 + hh] + ad_[d * 4 + hh]);
    atomicMax(&mkey[d * 4 + hh], fkey(lg));
  }
}

__global__ void decode_m_kernel(const unsigned* __restrict__ mkey, float* __restrict__ m,
                                float* __restrict__ z, int n) {
  int i = blockIdx.x * blockDim.x + threadIdx.x;
  if (i < n) { m[i] = fdecode(mkey[i]); z[i] = 0.f; }
}

// edge pass 2: segment sum of exp(logit - m[dst])
__global__ void edge_sum_kernel(const int* __restrict__ ei, int E, int N,
                                const float* __restrict__ as_, const float* __restrict__ ad_,
                                const float* __restrict__ m, float* __restrict__ z) {
  int e = blockIdx.x * blockDim.x + threadIdx.x;
  if (e >= E + N) return;
  int s = (e < E) ? ei[e] : (e - E);
  int d = (e < E) ? ei[E + e] : (e - E);
#pragma unroll
  for (int hh = 0; hh < 4; hh++) {
    float lg = lrelu02(as_[s * 4 + hh] + ad_[d * 4 + hh]);
    atomicAdd(&z[d * 4 + hh], expf(lg - m[d * 4 + hh]));
  }
}

// initialize aggregation buffer to bias (HC=256 -> col = i & 255)
__global__ void init_bias_kernel(float* __restrict__ out, const float* __restrict__ b, int total) {
  int i = blockIdx.x * blockDim.x + threadIdx.x;
  if (i < total) out[i] = b[i & 255];
}

// edge pass 3: out[dst] += h[src] * alpha ; one wave per edge, float4 per lane
__global__ __launch_bounds__(256) void edge_agg_kernel(const int* __restrict__ ei, int E, int N,
                                                       const float* __restrict__ as_,
                                                       const float* __restrict__ ad_,
                                                       const float* __restrict__ m,
                                                       const float* __restrict__ z,
                                                       const float* __restrict__ hsrc,
                                                       float* __restrict__ out) {
  int wid = (blockIdx.x * blockDim.x + threadIdx.x) >> 5;
  int l = threadIdx.x & 31;
  if (wid >= E + N) return;
  int s = (wid < E) ? ei[wid] : (wid - E);
  int d = (wid < E) ? ei[E + wid] : (wid - E);
  // lane's float4 at cols [4l,4l+3] -> head hA; at cols [128+4l ...] -> head hB
  int hA = l >> 4;
  int hB = 2 + (l >> 4);
  float lgA = lrelu02(as_[s * 4 + hA] + ad_[d * 4 + hA]);
  float lgB = lrelu02(as_[s * 4 + hB] + ad_[d * 4 + hB]);
  float aA = expf(lgA - m[d * 4 + hA]) / (z[d * 4 + hA] + 1e-16f);
  float aB = expf(lgB - m[d * 4 + hB]) / (z[d * 4 + hB] + 1e-16f);
  const float4* hs = (const float4*)(hsrc + (size_t)s * HC);
  float4 vA = hs[l];
  float4 vB = hs[l + 32];
  float* o = out + (size_t)d * HC;
  atomicAdd(&o[l * 4 + 0], vA.x * aA);
  atomicAdd(&o[l * 4 + 1], vA.y * aA);
  atomicAdd(&o[l * 4 + 2], vA.z * aA);
  atomicAdd(&o[l * 4 + 3], vA.w * aA);
  atomicAdd(&o[128 + l * 4 + 0], vB.x * aB);
  atomicAdd(&o[128 + l * 4 + 1], vB.y * aB);
  atomicAdd(&o[128 + l * 4 + 2], vB.z * aB);
  atomicAdd(&o[128 + l * 4 + 3], vB.w * aB);
}

// ELU in place; optionally emit f16 copy (input to next layer's GEMM)
__global__ void elu_kernel(float* __restrict__ x, _Float16* __restrict__ x16, int n) {
  int i = blockIdx.x * blockDim.x + threadIdx.x;
  if (i >= n) return;
  float v = x[i];
  float e = v > 0.f ? v : (expf(v) - 1.f);
  x[i] = e;
  if (x16) x16[i] = (_Float16)e;
}

// MLP head: relu(h @ lw1 + lb1) @ lw2 + lb2 -> log_softmax (2 classes)
__global__ __launch_bounds__(64) void mlp_head_kernel(const float* __restrict__ hin,
                                                      const float* __restrict__ lw1,
                                                      const float* __restrict__ lb1,
                                                      const float* __restrict__ lw2,
                                                      const float* __restrict__ lb2,
                                                      float* __restrict__ out) {
  __shared__ float row[HC];
  __shared__ float mid[OUTC];
  int n = blockIdx.x;
  int t = threadIdx.x;
  for (int i = t; i < HC; i += 64) row[i] = hin[(size_t)n * HC + i];
  __syncthreads();
  float acc = lb1[t];
  for (int k = 0; k < HC; k++) acc += row[k] * lw1[(size_t)k * OUTC + t];
  mid[t] = acc > 0.f ? acc : 0.f;
  __syncthreads();
  if (t == 0) {
    float o0 = lb2[0], o1 = lb2[1];
    for (int k = 0; k < OUTC; k++) { o0 += mid[k] * lw2[2 * k]; o1 += mid[k] * lw2[2 * k + 1]; }
    float mx = fmaxf(o0, o1);
    float lse = mx + logf(expf(o0 - mx) + expf(o1 - mx));
    out[(size_t)n * 2]     = o0 - lse;
    out[(size_t)n * 2 + 1] = o1 - lse;
  }
}

extern "C" void kernel_launch(void* const* d_in, const int* in_sizes, int n_in,
                              void* d_out, int out_size, void* d_ws, size_t ws_size,
                              hipStream_t stream) {
  (void)n_in; (void)out_size; (void)ws_size;
  const float* x   = (const float*)d_in[0];
  const int*   ei  = (const int*)d_in[1];
  const float* W1  = (const float*)d_in[2];
  const float* as1 = (const float*)d_in[3];
  const float* ad1 = (const float*)d_in[4];
  const float* b1  = (const float*)d_in[5];
  const float* W2  = (const float*)d_in[6];
  const float* as2 = (const float*)d_in[7];
  const float* ad2 = (const float*)d_in[8];
  const float* b2  = (const float*)d_in[9];
  const float* lw1 = (const float*)d_in[10];
  const float* lb1 = (const float*)d_in[11];
  const float* lw2 = (const float*)d_in[12];
  const float* lb2 = (const float*)d_in[13];
  float* out = (float*)d_out;

  const int F = 128;
  const int N = in_sizes[0] / F;
  const int E = in_sizes[1] / 2;
  const int ET = E + N;  // edges incl. self-loops

  // bump allocator over workspace
  char* p = (char*)d_ws;
  auto alloc = [&](size_t bytes) -> void* {
    void* r = (void*)p;
    p += (bytes + 255) & ~(size_t)255;
    return r;
  };
  _Float16* xh   = (_Float16*)alloc((size_t)N * F * sizeof(_Float16));
  unsigned* w1p  = (unsigned*)alloc((size_t)(F / 32) * 16 * 256 * 4);
  unsigned* w2p  = (unsigned*)alloc((size_t)(HC / 32) * 16 * 256 * 4);
  float*    h    = (float*)alloc((size_t)N * HC * 4);       // GEMM output (both layers)
  float*    agg  = (float*)alloc((size_t)N * HC * 4);       // aggregation output (both layers)
  _Float16* hact = (_Float16*)alloc((size_t)N * HC * 2);    // layer-1 activation, f16
  float*    asn  = (float*)alloc((size_t)N * 4 * 4);
  float*    adn  = (float*)alloc((size_t)N * 4 * 4);
  unsigned* mkey = (unsigned*)alloc((size_t)N * 4 * 4);
  float*    m    = (float*)alloc((size_t)N * 4 * 4);
  float*    z    = (float*)alloc((size_t)N * 4 * 4);

  const int B = 256;
  f32_to_f16_kernel<<<(N * F + B - 1) / B, B, 0, stream>>>(x, xh, N * F);
  int npk1 = (F / 32) * 16 * 256;
  int npk2 = (HC / 32) * 16 * 256;
  pack_b_kernel<<<(npk1 + B - 1) / B, B, 0, stream>>>(W1, w1p, F);
  pack_b_kernel<<<(npk2 + B - 1) / B, B, 0, stream>>>(W2, w2p, HC);

  int nTiles = (N / 16) * 16;
  int gemmBlocks = (nTiles * 32 + B - 1) / B;
  int nhc = N * HC;

  for (int layer = 0; layer < 2; layer++) {
    const _Float16* Ain  = layer ? hact : xh;
    const int       K    = layer ? HC : F;
    const unsigned* Bp   = layer ? w2p : w1p;
    const float*    asw  = layer ? as2 : as1;
    const float*    adw  = layer ? ad2 : ad1;
    const float*    bias = layer ? b2 : b1;

    gemm_wmma_kernel<<<gemmBlocks, B, 0, stream>>>(Ain, Bp, h, N, K);
    att_score_kernel<<<N, 128, 0, stream>>>(h, asw, adw, asn, adn);
    init_mkey_kernel<<<(N * 4 + B - 1) / B, B, 0, stream>>>(mkey, N * 4);
    edge_max_kernel<<<(ET + B - 1) / B, B, 0, stream>>>(ei, E, N, asn, adn, mkey);
    decode_m_kernel<<<(N * 4 + B - 1) / B, B, 0, stream>>>(mkey, m, z, N * 4);
    edge_sum_kernel<<<(ET + B - 1) / B, B, 0, stream>>>(ei, E, N, asn, adn, m, z);
    init_bias_kernel<<<(nhc + B - 1) / B, B, 0, stream>>>(agg, bias, nhc);
    edge_agg_kernel<<<((long long)ET * 32 + B - 1) / B, B, 0, stream>>>(ei, E, N, asn, adn, m, z, h, agg);
    elu_kernel<<<(nhc + B - 1) / B, B, 0, stream>>>(agg, layer ? (_Float16*)nullptr : hact, nhc);
  }

  mlp_head_kernel<<<N, 64, 0, stream>>>(agg, lw1, lb1, lw2, lb2, out);
}
